// LatentProcessor_78434692760025
// MI455X (gfx1250) — matrix operand order repaired and checked by reading, not compiled, not run on hardware
//
#include <hip/hip_runtime.h>
#include <hip/hip_bf16.h>
#include <math.h>

// ---------------- problem dimensions (fixed by the reference) ----------------
#define B_        8
#define T_        1024
#define IN_       512
#define D_        1024
#define I_        2048
#define N_        64
#define H_        16
#define P_        128      // I_/H_
#define KC_       3        // conv kernel
#define L_        4
#define OUT_      512
#define CB_       1024
#define CONV_DIM  (I_ + 2 * N_)        // 2176
#define PROJ      (I_ + CONV_DIM + H_) // 4240
#define M_ROWS    (B_ * T_)            // 8192

// ---------------- WMMA types (CDNA5 / gfx1250, wave32) ----------------
typedef __attribute__((ext_vector_type(16))) __bf16 v16bf;
typedef __attribute__((ext_vector_type(8)))  __bf16 v8bf;
typedef __attribute__((ext_vector_type(8)))  float  v8f;

// ---------------- generic WMMA GEMM: C = A[MxK] * W[KxN] (+bias) (+resid) ----
// Block tile: 128(M) x 64(N), K-step 32. 256 threads = 8 waves.
// LDS tiles are stored in *fragment order*: each lane's v16bf fragment is 32
// contiguous bytes -> loads lower to ds_load_b128 pairs instead of 16x u16.
#define TM 128
#define TN 64
#define TK 32

__global__ __launch_bounds__(256)
void wmma_gemm_kernel(const float* __restrict__ A, const float* __restrict__ W,
                      const float* __restrict__ bias, const float* __restrict__ resid,
                      float* __restrict__ C, int M, int N, int K)
{
    // fragment-order LDS: Af[rb][lane][j], Bf[cb][lane][j], j = 0..15 bf16
    __shared__ __align__(32) __bf16 Af[8 * 32 * 16];   // 8 KB
    __shared__ __align__(32) __bf16 Bf[4 * 32 * 16];   // 4 KB

    const int tid   = threadIdx.x;
    const int wave  = tid >> 5;
    const int lane  = tid & 31;
    const int lhalf = lane >> 4;
    const int l16   = lane & 15;

    const int m0 = blockIdx.y * TM;     // M is always a multiple of TM here
    const int n0 = blockIdx.x * TN;

    // B staging assignment (uniform across K loop): thread -> (n, k-group)
    const int bn  = tid >> 2;            // 0..63
    const int bkg = tid & 3;             // k-group of 8
    const int bgn = n0 + bn;
    const bool bn_ok = (bgn < N);
    {
        // precompute B LDS slot
    }
    const int bcb = bn >> 4, bnn = bn & 15;
    __bf16* const bdst = &Bf[(((bcb << 5) + (bnn + ((bkg & 1) << 4))) * 16) + ((bkg >> 1) << 3)];

    v8f acc[4];
    #pragma unroll
    for (int i = 0; i < 4; ++i) acc[i] = (v8f){};

    for (int k0 = 0; k0 < K; k0 += TK) {
        // --- stage A tile (128 rows x 32 k) in fragment order ---
        // group = (row, kg): 8 consecutive k -> contiguous j -> one 16B LDS store
        #pragma unroll
        for (int i = 0; i < 2; ++i) {
            int g  = tid + 256 * i;          // 0..511
            int r  = g >> 2;                 // 0..127
            int kg = g & 3;                  // k-group of 8
            const float4* ap = (const float4*)&A[(size_t)(m0 + r) * K + (k0 + (kg << 3))];
            float4 f0 = ap[0], f1 = ap[1];
            v8bf pk;
            pk[0] = (__bf16)f0.x; pk[1] = (__bf16)f0.y;
            pk[2] = (__bf16)f0.z; pk[3] = (__bf16)f0.w;
            pk[4] = (__bf16)f1.x; pk[5] = (__bf16)f1.y;
            pk[6] = (__bf16)f1.z; pk[7] = (__bf16)f1.w;
            int rb = r >> 4, m = r & 15;
            int dl = m + ((kg & 1) << 4);    // dest lane
            int j0 = (kg >> 1) << 3;         // dest j base
            *(v8bf*)&Af[((rb << 5) + dl) * 16 + j0] = pk;
        }
        // --- stage B tile (32 k x 64 n) in fragment order ---
        // guard hoisted: single branch, 8 clause-able strided loads inside
        {
            v8bf pk;
            if (bn_ok) {
                const float* wp = &W[(size_t)(k0 + (bkg << 3)) * N + bgn];
                #pragma unroll
                for (int q = 0; q < 8; ++q)
                    pk[q] = (__bf16)wp[(size_t)q * N];
            } else {
                #pragma unroll
                for (int q = 0; q < 8; ++q)
                    pk[q] = (__bf16)0.0f;
            }
            *(v8bf*)bdst = pk;
        }
        // prefetch next K tile (global_prefetch_b8)
        if (k0 + TK < K) {
            __builtin_prefetch(&A[(size_t)(m0 + (tid & 127)) * K + (k0 + TK)], 0, 1);
            if (tid < TK && bn_ok)
                __builtin_prefetch(&W[(size_t)(k0 + TK + tid) * N + n0 + (l16 << 2)], 0, 1);
        }
        __syncthreads();

        // --- fragments: one v16bf (2x ds_load_b128) each ---
        const v16bf af = *(const v16bf*)&Af[((wave << 5) + lane) * 16];
        #pragma unroll
        for (int nb = 0; nb < 4; ++nb) {
            const v16bf bf = *(const v16bf*)&Bf[((nb << 5) + lane) * 16];
            acc[nb] = __builtin_amdgcn_wmma_f32_16x16x32_bf16(
                false, af, false, bf, (short)0, acc[nb], false, false);
        }
        __syncthreads();
    }

    // --- epilogue: f32 C layout: row = 16*wave + r + 8*lhalf, col = 16*nb + l16
    #pragma unroll
    for (int nb = 0; nb < 4; ++nb) {
        #pragma unroll
        for (int r = 0; r < 8; ++r) {
            int row = m0 + (wave << 4) + r + (lhalf << 3);
            int col = n0 + (nb << 4) + l16;
            if (col < N) {
                float v = acc[nb][r];
                if (bias)  v += bias[col];
                if (resid) v += resid[(size_t)row * N + col];   // elementwise-aligned: in-place safe
                C[(size_t)row * N + col] = v;
            }
        }
    }
}

// ---------------- RMSNorm (optionally fused with silu(gate)) ----------------
__global__ __launch_bounds__(256)
void rmsnorm_kernel(const float* __restrict__ x, int xstride,
                    const float* __restrict__ gate, int gstride,
                    const float* __restrict__ w,
                    float* __restrict__ out, int ostride, int ncols)
{
    __shared__ float red[8];
    const int row = blockIdx.x;
    const float* xr = x + (size_t)row * xstride;
    const float* gr = gate ? gate + (size_t)row * gstride : nullptr;

    float ss = 0.f;
    for (int c = threadIdx.x; c < ncols; c += 256) {
        float v = xr[c];
        if (gr) { float g = gr[c]; v *= g / (1.f + expf(-g)); }
        ss += v * v;
    }
    #pragma unroll
    for (int off = 16; off > 0; off >>= 1) ss += __shfl_down(ss, off, 32);
    if ((threadIdx.x & 31) == 0) red[threadIdx.x >> 5] = ss;
    __syncthreads();
    if (threadIdx.x == 0) {
        float t = 0.f;
        #pragma unroll
        for (int i = 0; i < 8; ++i) t += red[i];
        red[0] = rsqrtf(t / (float)ncols + 1e-6f);
    }
    __syncthreads();
    const float scale = red[0];
    float* orow = out + (size_t)row * ostride;
    for (int c = threadIdx.x; c < ncols; c += 256) {
        float v = xr[c];
        if (gr) { float g = gr[c]; v *= g / (1.f + expf(-g)); }
        orow[c] = v * scale * w[c];   // in-place (out==x) safe: read-then-write per element
    }
}

// ---------------- causal depthwise conv (K=3) + SiLU ----------------
// hbc points at proj[:, :, I_] (row stride PROJ); out is dense [B*T, CONV_DIM].
__global__ __launch_bounds__(256)
void conv_silu_kernel(const float* __restrict__ hbc, const float* __restrict__ cw,
                      const float* __restrict__ cb, float* __restrict__ out)
{
    const long i = (long)blockIdx.x * 256 + threadIdx.x;
    const long total = (long)M_ROWS * CONV_DIM;
    if (i >= total) return;
    const int c = (int)(i % CONV_DIM);
    const int t = (int)((i / CONV_DIM) % T_);
    const int b = (int)(i / ((long)CONV_DIM * T_));
    float acc = cb[c];
    #pragma unroll
    for (int k = 0; k < KC_; ++k) {
        int tt = t - (KC_ - 1) + k;
        if (tt >= 0)
            acc += cw[k * CONV_DIM + c] * hbc[(size_t)(b * T_ + tt) * PROJ + c];
    }
    out[i] = acc / (1.f + expf(-acc));
}

// ---------------- dt = softplus(proj_dt + dt_bias) ----------------
__global__ __launch_bounds__(256)
void dt_softplus_kernel(const float* __restrict__ proj_dt, const float* __restrict__ dtb,
                        float* __restrict__ out)
{
    const int i = blockIdx.x * 256 + threadIdx.x;
    if (i >= M_ROWS * H_) return;
    const int h = i % H_;
    const int r = i / H_;
    float v = proj_dt[(size_t)r * PROJ + h] + dtb[h];
    out[i] = (v > 20.f) ? v : log1pf(expf(v));
}

// ---------------- SSM selective scan ----------------
// One block per (b,h); 128 threads, thread p owns state[p, 0..63] in registers.
__global__ __launch_bounds__(128)
void ssm_scan_kernel(const float* __restrict__ conv,  // [B*T, CONV_DIM]
                     const float* __restrict__ dt,    // [B*T, H_]
                     const float* __restrict__ A_log, // [H_] (layer slice)
                     const float* __restrict__ Dv,    // [H_]
                     float* __restrict__ y)           // [B*T, I_]
{
    __shared__ float Bs[N_];
    __shared__ float Cs[N_];
    const int b = blockIdx.x / H_;
    const int h = blockIdx.x % H_;
    const int p = threadIdx.x;
    const float a = -expf(A_log[h]);
    const float d = Dv[h];

    float st[N_];
    #pragma unroll
    for (int n = 0; n < N_; ++n) st[n] = 0.f;

    for (int t = 0; t < T_; ++t) {
        const size_t row = (size_t)(b * T_ + t);
        if (threadIdx.x < 2 * N_)   // threads 0-63 -> B, 64-127 -> C (contiguous slices)
            ((threadIdx.x < N_) ? Bs[threadIdx.x] : Cs[threadIdx.x - N_]) =
                conv[row * CONV_DIM + I_ + threadIdx.x];
        __syncthreads();

        const float dtv = dt[row * H_ + h];
        const float xv  = conv[row * CONV_DIM + h * P_ + p];
        const float dA  = expf(dtv * a);
        const float dtx = dtv * xv;
        float acc = 0.f;
        #pragma unroll
        for (int n = 0; n < N_; ++n) {
            st[n] = fmaf(st[n], dA, dtx * Bs[n]);
            acc   = fmaf(st[n], Cs[n], acc);
        }
        y[row * I_ + h * P_ + p] = acc + d * xv;
        __syncthreads();
    }
}

// ---------------- host-side orchestration ----------------
static inline void launch_gemm(const float* A, const float* W, const float* bias,
                               const float* resid, float* C, int M, int N, int K,
                               hipStream_t stream)
{
    dim3 grid((N + TN - 1) / TN, M / TM);
    wmma_gemm_kernel<<<grid, dim3(256), 0, stream>>>(A, W, bias, resid, C, M, N, K);
}

extern "C" void kernel_launch(void* const* d_in, const int* in_sizes, int n_in,
                              void* d_out, int out_size, void* d_ws, size_t ws_size,
                              hipStream_t stream)
{
    const float* x        = (const float*)d_in[0];
    const float* in_w     = (const float*)d_in[1];
    const float* in_b     = (const float*)d_in[2];
    const float* norm_w   = (const float*)d_in[3];
    const float* mix_in_w = (const float*)d_in[4];
    const float* conv_w   = (const float*)d_in[5];
    const float* conv_b   = (const float*)d_in[6];
    const float* dt_bias  = (const float*)d_in[7];
    const float* A_log    = (const float*)d_in[8];
    const float* Dv       = (const float*)d_in[9];
    const float* gnorm_w  = (const float*)d_in[10];
    const float* mix_out_w= (const float*)d_in[11];
    const float* out_w    = (const float*)d_in[12];
    const float* out_b    = (const float*)d_in[13];
    const float* code_w   = (const float*)d_in[14];
    const float* code_b   = (const float*)d_in[15];
    (void)in_sizes; (void)n_in; (void)out_size; (void)ws_size;

    // workspace layout (fp32)
    float* ws   = (float*)d_ws;
    float* h    = ws;                                  // [M_ROWS, D_]
    float* hn   = h    + (size_t)M_ROWS * D_;          // [M_ROWS, D_]
    float* proj = hn   + (size_t)M_ROWS * D_;          // [M_ROWS, PROJ]
    float* conv = proj + (size_t)M_ROWS * PROJ;        // [M_ROWS, CONV_DIM]
    float* dts  = conv + (size_t)M_ROWS * CONV_DIM;    // [M_ROWS, H_]
    float* yb   = dts  + (size_t)M_ROWS * H_;          // [M_ROWS, I_]

    // h = x @ in_w + in_b
    launch_gemm(x, in_w, in_b, nullptr, h, M_ROWS, D_, IN_, stream);

    for (int l = 0; l < L_; ++l) {
        // hn = rmsnorm(h)
        rmsnorm_kernel<<<M_ROWS, 256, 0, stream>>>(h, D_, nullptr, 0,
                                                   norm_w + (size_t)l * D_, hn, D_, D_);
        // proj = hn @ mix_in_w[l]
        launch_gemm(hn, mix_in_w + (size_t)l * D_ * PROJ, nullptr, nullptr,
                    proj, M_ROWS, PROJ, D_, stream);
        // conv = silu(depthwise_conv(proj[:, I_:I_+CONV_DIM]) + conv_b)
        {
            long total = (long)M_ROWS * CONV_DIM;
            conv_silu_kernel<<<(unsigned)((total + 255) / 256), 256, 0, stream>>>(
                proj + I_, conv_w + (size_t)l * KC_ * CONV_DIM,
                conv_b + (size_t)l * CONV_DIM, conv);
        }
        // dts = softplus(proj[:, I_+CONV_DIM:] + dt_bias)
        dt_softplus_kernel<<<(M_ROWS * H_ + 255) / 256, 256, 0, stream>>>(
            proj + I_ + CONV_DIM, dt_bias + (size_t)l * H_, dts);
        // yb = scan(conv, dts) + D*x
        ssm_scan_kernel<<<B_ * H_, 128, 0, stream>>>(
            conv, dts, A_log + (size_t)l * H_, Dv + (size_t)l * H_, yb);
        // yb = rmsnorm(yb * silu(gate)) in place; gate = proj[:, :I_] (stride PROJ)
        rmsnorm_kernel<<<M_ROWS, 256, 0, stream>>>(yb, I_, proj, PROJ,
                                                   gnorm_w + (size_t)l * I_, yb, I_, I_);
        // h = h + yb @ mix_out_w[l]  (in-place residual epilogue)
        launch_gemm(yb, mix_out_w + (size_t)l * I_ * D_, nullptr, h,
                    h, M_ROWS, D_, I_, stream);
    }

    float* out0 = (float*)d_out;
    launch_gemm(h, out_w, out_b, nullptr, out0, M_ROWS, OUT_, D_, stream);
    launch_gemm(h, code_w, code_b, nullptr, out0 + (size_t)M_ROWS * OUT_,
                M_ROWS, CB_, D_, stream);
}
